// ScaledDotProductionAttention_38508676776685
// MI455X (gfx1250) — matrix operand order repaired and checked by reading, not compiled, not run on hardware
//
#include <hip/hip_runtime.h>
#include <stdint.h>

// Flash-style attention on gfx1250 (wave32, WMMA f16, f32 accum).
// B=4,H=16,S=2048,D=64. Computes S^T = K.Q^T so softmax stats are per-lane
// scalars and P feeds the P.V WMMA straight from registers (no LDS round-trip).
// Softmax runs in base-2 with scale*log2(e) folded into the Q fragments.

typedef _Float16 f16;
typedef __attribute__((ext_vector_type(16))) _Float16 v16h;
typedef __attribute__((ext_vector_type(8)))  _Float16 v8h;
typedef __attribute__((ext_vector_type(2)))  _Float16 h2;
typedef __attribute__((ext_vector_type(2)))  __fp16   fp16x2;
typedef __attribute__((ext_vector_type(8)))  float    v8f;

#define SLEN   2048
#define DHEAD  64
#define NBH    64                     // B*H
#define TK     64                     // key tile
#define NT     (SLEN / TK)
#define KSTR   72                     // padded f16 row stride (16B-aligned rows)
#define QTILE  128                    // 8 waves * 16 query rows
#define SCLL2E 0.18033688011116016f   // (1/sqrt(64)) * log2(e)
#define MASKC  (-2.0e9f)              // masked score in log2 domain

// v_cvt_pk_rtz_f16_f32 wrapper (builtin returns __fp16x2; bit-cast to h2).
__device__ __forceinline__ h2 pkrtz(float x, float y) {
    fp16x2 t = __builtin_amdgcn_cvt_pkrtz(x, y);
    return __builtin_bit_cast(h2, t);
}

// A-fragment (16x32 f16): elems 0..7 at p[0..7], elems 8..15 at p[16..23].
__device__ __forceinline__ v16h ld_frag_a(const f16* p) {
    v8h lo = *(const v8h*)(p);
    v8h hi = *(const v8h*)(p + 16);
    v16h r;
#pragma unroll
    for (int i = 0; i < 8; ++i) { r[i] = lo[i]; r[i + 8] = hi[i]; }
    return r;
}
// B-fragment (32x16 f16): 16 contiguous K elements per lane.
__device__ __forceinline__ v16h ld_frag_b(const f16* p) {
    v8h lo = *(const v8h*)(p);
    v8h hi = *(const v8h*)(p + 8);
    v16h r;
#pragma unroll
    for (int i = 0; i < 8; ++i) { r[i] = lo[i]; r[i + 8] = hi[i]; }
    return r;
}

__global__ __launch_bounds__(256)
void ScaledDotProductionAttention_38508676776685_kernel(
    const float* __restrict__ Q,
    const float* __restrict__ K,
    const float* __restrict__ V,
    const uint8_t* __restrict__ mask,
    float* __restrict__ out)
{
    __shared__ __align__(16) f16 sK[2][TK * KSTR];      // [key][d], double-buffered
    __shared__ __align__(16) f16 sVt[2][DHEAD * KSTR];  // [d][key], double-buffered

    const int tid   = threadIdx.x;
    const int wave  = tid >> 5;
    const int lane  = tid & 31;
    const int lhalf = lane >> 4;
    const int lcol  = lane & 15;

    const int bh    = blockIdx.y;
    const int qbase = blockIdx.x * QTILE + wave * 16;

    const size_t headOff = (size_t)bh * SLEN * DHEAD;
    const float* Qh = Q + headOff;
    const float* Kh = K + headOff;
    const float* Vh = V + headOff;

    // Staging coordinates: each thread covers rows {r0,r0+1}, cols c0..c0+7.
    const int r0 = (tid >> 3) << 1;
    const int c0 = (tid & 7) * 8;

    // ---- Q B-fragments, scaled by (1/sqrt(d))*log2(e), resident in regs ----
    v16h qb[2];
    {
        const float* qp = Qh + (size_t)(qbase + lcol) * DHEAD;
#pragma unroll
        for (int ch = 0; ch < 2; ++ch) {
            v16h b;
#pragma unroll
            for (int i = 0; i < 16; ++i)
                b[i] = (f16)(qp[ch * 32 + 8 * lhalf + (i & 7) + ((i >> 3) << 4)] * SCLL2E);
            qb[ch] = b;
        }
    }

    // Online softmax state: this lane owns query row (qbase+lcol).
    float mrun = -1e30f, lrun = 0.0f;
    v8f oacc[4];
#pragma unroll
    for (int dt = 0; dt < 4; ++dt) oacc[dt] = (v8f){};

    const uint8_t* mrow = mask + ((size_t)bh * SLEN + qbase + lcol) * SLEN;

    // ---- pipelined staging: regs for next tile's K/V (f16 packed) ----
    h2 stk[8], stv[8];
    auto stage_load = [&](int kt) {
        const int kb = kt * TK;
        const float* kp = Kh + (size_t)(kb + r0) * DHEAD + c0;
        const float* vp = Vh + (size_t)(kb + r0) * DHEAD + c0;
        float4 a, b;
        a = *(const float4*)kp;           b = *(const float4*)(kp + 4);
        stk[0] = pkrtz(a.x, a.y);  stk[1] = pkrtz(a.z, a.w);
        stk[2] = pkrtz(b.x, b.y);  stk[3] = pkrtz(b.z, b.w);
        a = *(const float4*)(kp + DHEAD); b = *(const float4*)(kp + DHEAD + 4);
        stk[4] = pkrtz(a.x, a.y);  stk[5] = pkrtz(a.z, a.w);
        stk[6] = pkrtz(b.x, b.y);  stk[7] = pkrtz(b.z, b.w);
        a = *(const float4*)vp;           b = *(const float4*)(vp + 4);
        stv[0] = pkrtz(a.x, a.y);  stv[1] = pkrtz(a.z, a.w);
        stv[2] = pkrtz(b.x, b.y);  stv[3] = pkrtz(b.z, b.w);
        a = *(const float4*)(vp + DHEAD); b = *(const float4*)(vp + DHEAD + 4);
        stv[4] = pkrtz(a.x, a.y);  stv[5] = pkrtz(a.z, a.w);
        stv[6] = pkrtz(b.x, b.y);  stv[7] = pkrtz(b.z, b.w);
    };
    auto stage_store = [&](int buf) {
        f16* kb = sK[buf];
        f16* vb = sVt[buf];
        *(v8h*)&kb[r0 * KSTR + c0] =
            (v8h){stk[0][0], stk[0][1], stk[1][0], stk[1][1],
                  stk[2][0], stk[2][1], stk[3][0], stk[3][1]};
        *(v8h*)&kb[(r0 + 1) * KSTR + c0] =
            (v8h){stk[4][0], stk[4][1], stk[5][0], stk[5][1],
                  stk[6][0], stk[6][1], stk[7][0], stk[7][1]};
#pragma unroll
        for (int i = 0; i < 8; ++i) {
            h2 t = (h2){stv[i >> 1][i & 1], stv[4 + (i >> 1)][i & 1]};
            *(h2*)&vb[(c0 + i) * KSTR + r0] = t;
        }
    };

    stage_load(0);
    stage_store(0);
    __syncthreads();

    for (int kt = 0; kt < NT; ++kt) {
        const int  cur  = kt & 1;
        const bool more = (kt + 1 < NT);
        if (more) stage_load(kt + 1);   // global loads overlap this tile's compute

        const f16* sKb = sK[cur];
        const f16* sVb = sVt[cur];
        const size_t kbo = (size_t)kt * TK;

        // Mask bytes for this lane's query: 4 x 8 contiguous keys.
        uint2 mw[4];
#pragma unroll
        for (int ch = 0; ch < 2; ++ch) {
            mw[2 * ch]     = *(const uint2*)(mrow + kbo + ch * 32 + 8 * lhalf);
            mw[2 * ch + 1] = *(const uint2*)(mrow + kbo + ch * 32 + 16 + 8 * lhalf);
        }

        // ---- S^T = K . Q^T : lane holds query lcol, keys on VGPR index ----
        v8f sc[4];
#pragma unroll
        for (int j = 0; j < 4; ++j) {
            v8f c = (v8f){};
#pragma unroll
            for (int ch = 0; ch < 2; ++ch) {
                v16h a = ld_frag_a(&sKb[(j * 16 + lcol) * KSTR + ch * 32 + 8 * lhalf]);
                c = __builtin_amdgcn_wmma_f32_16x16x32_f16(
                        false, a, false, qb[ch], (short)0, c, false, false);
            }
            sc[j] = c;
        }

        // ---- mask (byte != 0 -> MASKC) ----
#pragma unroll
        for (int j = 0; j < 4; ++j) {
#pragma unroll
            for (int v = 0; v < 8; ++v) {
                const uint32_t w  = (v < 4) ? mw[j].x : mw[j].y;
                const uint32_t mb = (w >> (8 * (v & 3))) & 0xffu;
                sc[j][v] = mb ? MASKC : sc[j][v];
            }
        }

        // ---- online softmax (scalar per lane; one cross-lane exchange) ----
        float mx = sc[0][0];
#pragma unroll
        for (int j = 0; j < 4; ++j)
#pragma unroll
            for (int v = 0; v < 8; ++v) mx = fmaxf(mx, sc[j][v]);
        mx = fmaxf(mx, __shfl_xor(mx, 16, 32));
        const float mn   = fmaxf(mrun, mx);
        const float corr = __builtin_exp2f(mrun - mn);
        mrun = mn;

        // Broadcast corr to the output-fragment row owners and rescale O.
        float corrv[8];
#pragma unroll
        for (int v = 0; v < 8; ++v) corrv[v] = __shfl(corr, 8 * lhalf + v, 16);
#pragma unroll
        for (int dt = 0; dt < 4; ++dt)
#pragma unroll
            for (int v = 0; v < 8; ++v) oacc[dt][v] *= corrv[v];

        // P = 2^(s - m); accumulate row sum.
        float ps = 0.0f;
#pragma unroll
        for (int j = 0; j < 4; ++j)
#pragma unroll
            for (int v = 0; v < 8; ++v) {
                const float p = __builtin_exp2f(sc[j][v] - mn);
                sc[j][v] = p;
                ps += p;
            }
        ps += __shfl_xor(ps, 16, 32);
        lrun = fmaf(lrun, corr, ps);

        // ---- P A-fragments straight from registers (packed cvt) ----
        v16h pa[2];
#pragma unroll
        for (int ch = 0; ch < 2; ++ch) {
            v16h a;
#pragma unroll
            for (int t = 0; t < 4; ++t) {
                h2 x = pkrtz(sc[2 * ch][2 * t], sc[2 * ch][2 * t + 1]);
                a[2 * t] = x[0]; a[2 * t + 1] = x[1];
            }
#pragma unroll
            for (int t = 0; t < 4; ++t) {
                h2 x = pkrtz(sc[2 * ch + 1][2 * t], sc[2 * ch + 1][2 * t + 1]);
                a[8 + 2 * t] = x[0]; a[8 + 2 * t + 1] = x[1];
            }
            pa[ch] = a;
        }

        // ---- O += P . V ----
#pragma unroll
        for (int dt = 0; dt < 4; ++dt) {
            v8f c = oacc[dt];
#pragma unroll
            for (int ch = 0; ch < 2; ++ch) {
                v16h b = ld_frag_a(&sVb[(dt * 16 + lcol) * KSTR + ch * 32 + 8 * lhalf]);
                c = __builtin_amdgcn_wmma_f32_16x16x32_f16(
                        false, pa[ch], false, b, (short)0, c, false, false);
            }
            oacc[dt] = c;
        }

        if (more) stage_store(1 - cur);  // stores land after loads returned
        __syncthreads();
    }

    // ---- normalize (broadcast 1/l to row owners) and store f32 ----
    const float inv = 1.0f / lrun;
    float invv[8];
#pragma unroll
    for (int v = 0; v < 8; ++v) invv[v] = __shfl(inv, 8 * lhalf + v, 16);
    const int qm0 = qbase + 8 * lhalf;
#pragma unroll
    for (int v = 0; v < 8; ++v) {
        float* op = out + ((size_t)bh * SLEN + (qm0 + v)) * DHEAD + lcol;
#pragma unroll
        for (int dt = 0; dt < 4; ++dt)
            op[dt * 16] = oacc[dt][v] * invv[v];
    }
}

extern "C" void kernel_launch(void* const* d_in, const int* in_sizes, int n_in,
                              void* d_out, int out_size, void* d_ws, size_t ws_size,
                              hipStream_t stream) {
    const float*   Q    = (const float*)d_in[0];
    const float*   K    = (const float*)d_in[1];
    const float*   V    = (const float*)d_in[2];
    const uint8_t* mask = (const uint8_t*)d_in[3];
    float*         out  = (float*)d_out;

    dim3 grid(SLEN / QTILE, NBH); // (16, 64)
    dim3 block(256);
    ScaledDotProductionAttention_38508676776685_kernel<<<grid, block, 0, stream>>>(
        Q, K, V, mask, out);
}
